// SimpleLSTM_17746804867098
// MI455X (gfx1250) — matrix operand order repaired and checked by reading, not compiled, not run on hardware
//
#include <hip/hip_runtime.h>
#include <hip/hip_bf16.h>

typedef __attribute__((ext_vector_type(16))) __bf16 v16bf;
typedef __attribute__((ext_vector_type(8)))  __bf16 v8bf;
typedef __attribute__((ext_vector_type(8)))  float  v8f;

namespace lstmk {

constexpr int EMB = 512, HID = 512, OUTN = 2, BATCH = 32, SEQ = 2048;
constexpr int KDIM  = EMB + HID;      // 1024 (x | h fused)
constexpr int NGATE = 4 * HID;        // 2048 gate rows per layer
constexpr int NWG   = 32;             // 16 WGs per layer
constexpr int HCHUNK = 32;            // hidden units owned per WG
constexpr int NLOC   = 4 * HCHUNK;    // 128 gate columns per WG

// ---- workspace layout (bytes) ----
constexpr size_t OFF_WC   = 0;                                   // bf16 Wc[2][2048][1024]
constexpr size_t SZ_WC    = (size_t)2 * NGATE * KDIM * 2;        // 8 MiB
constexpr size_t OFF_BIAS = OFF_WC + SZ_WC;                      // f32 bias[2][2048]
constexpr size_t SZ_BIAS  = (size_t)2 * NGATE * 4;
constexpr size_t OFF_IN0  = OFF_BIAS + SZ_BIAS;                  // bf16 inbuf0[2][32][1024]
constexpr size_t SZ_INB   = (size_t)2 * BATCH * KDIM * 2;        // 128 KiB
constexpr size_t OFF_IN1  = OFF_IN0 + SZ_INB;                    // bf16 inbuf1[2][32][1024]
constexpr size_t OFF_H1F  = OFF_IN1 + SZ_INB;                    // f32 h1_final[32][512]
constexpr size_t SZ_H1F   = (size_t)BATCH * HID * 4;
constexpr size_t OFF_SYNC = OFF_H1F + SZ_H1F;                    // u32 cnt, u32 gen

// ---- LDS layout (bytes) ----
constexpr int WSL_BYTES  = 8 * 32 * 1024;        // 256 B-frags x 1024 B = 256 KiB
constexpr int GATE_BYTES = BATCH * NLOC * 4;     // 16 KiB  f32 gates[32][128]
constexpr int C_BYTES    = BATCH * HCHUNK * 4;   // 4 KiB   f32 c[32][32]
constexpr int SMEM_BYTES = WSL_BYTES + GATE_BYTES + C_BYTES;   // 276 KiB < 320 KiB

__device__ __forceinline__ unsigned short f2bf(float f) {
  unsigned u = __builtin_bit_cast(unsigned, f);
  unsigned r = u + 0x7FFFu + ((u >> 16) & 1u);   // round-to-nearest-even
  return (unsigned short)(r >> 16);
}
__device__ __forceinline__ float sigm(float x) {
  return 1.0f / (1.0f + __expf(-x));
}
__device__ __forceinline__ float tanh_fast(float x) {
  x = fminf(15.0f, fmaxf(-15.0f, x));
  float e = __expf(2.0f * x);
  return (e - 1.0f) / (e + 1.0f);
}
__device__ __forceinline__ v16bf cat16(v8bf lo, v8bf hi) {
  return __builtin_shufflevector(lo, hi, 0,1,2,3,4,5,6,7,8,9,10,11,12,13,14,15);
}

__device__ __forceinline__ void grid_sync(unsigned* cnt, unsigned* gen) {
  __syncthreads();
  if (threadIdx.x == 0) {
    __threadfence();
    unsigned g = __hip_atomic_load(gen, __ATOMIC_RELAXED, __HIP_MEMORY_SCOPE_AGENT);
    unsigned a = __hip_atomic_fetch_add(cnt, 1u, __ATOMIC_ACQ_REL, __HIP_MEMORY_SCOPE_AGENT);
    if (a == (unsigned)(NWG - 1)) {
      __hip_atomic_store(cnt, 0u, __ATOMIC_RELAXED, __HIP_MEMORY_SCOPE_AGENT);
      __hip_atomic_fetch_add(gen, 1u, __ATOMIC_RELEASE, __HIP_MEMORY_SCOPE_AGENT);
    } else {
      while (__hip_atomic_load(gen, __ATOMIC_ACQUIRE, __HIP_MEMORY_SCOPE_AGENT) == g)
        __builtin_amdgcn_s_sleep(2);
    }
    __threadfence();
  }
  __syncthreads();
}

// ---------------- prep: fuse + convert weights to bf16, combine biases ----------------
__global__ void prep_weights(const float* __restrict__ w_ih, const float* __restrict__ w_hh,
                             const float* __restrict__ b_ih, const float* __restrict__ b_hh,
                             unsigned short* __restrict__ Wc, float* __restrict__ bias) {
  int idx = blockIdx.x * 256 + threadIdx.x;          // 2*2048*1024 elements
  if (idx < 2 * NGATE * KDIM) {
    int k = idx & (KDIM - 1);
    int n = (idx >> 10) & (NGATE - 1);
    int l = idx >> 21;
    float v = (k < EMB) ? w_ih[((size_t)l * NGATE + n) * EMB + k]
                        : w_hh[((size_t)l * NGATE + n) * HID + (k - EMB)];
    Wc[idx] = f2bf(v);
  }
  if (idx < 2 * NGATE) bias[idx] = b_ih[idx] + b_hh[idx];
}

// ---------------- prep: zero activation buffers, load x(t=0), reset barrier ----------------
__global__ void prep_state(const int* __restrict__ tokens, const float* __restrict__ emb,
                           unsigned short* __restrict__ inb0, unsigned short* __restrict__ inb1,
                           unsigned* __restrict__ sync) {
  int idx = blockIdx.x * 256 + threadIdx.x;          // 65536 elements per buffer
  if (idx < 2) sync[idx] = 0u;
  if (idx < 2 * BATCH * KDIM) {
    inb1[idx] = 0;
    int par = idx >> 15;
    int b   = (idx >> 10) & 31;
    int k   = idx & (KDIM - 1);
    unsigned short v = 0;
    if (par == 0 && k < EMB) {
      int tok = tokens[(size_t)b * SEQ];             // t = 0
      v = f2bf(emb[(size_t)tok * EMB + k]);
    }
    inb0[idx] = v;
  }
}

// ---------------- persistent LSTM kernel: 32 WGs, 256 threads (8 waves) ----------------
__global__ void __launch_bounds__(256, 1)
lstm_persist(const int* __restrict__ tokens, const float* __restrict__ emb,
             const unsigned short* __restrict__ Wc, const float* __restrict__ bias,
             unsigned short* __restrict__ inb0, unsigned short* __restrict__ inb1,
             float* __restrict__ h1f, unsigned* __restrict__ sync,
             const float* __restrict__ fc_w, const float* __restrict__ fc_b,
             float* __restrict__ out) {
  extern __shared__ char smem[];
  unsigned short* wsl      = (unsigned short*)smem;                         // 256 KiB B-frags
  float*          gatesbuf = (float*)(smem + WSL_BYTES);                    // [32][128]
  float*          cbuf     = (float*)(smem + WSL_BYTES + GATE_BYTES);       // [32][32]

  const int wg    = blockIdx.x;
  const int layer = wg >> 4;
  const int wgl   = wg & 15;
  const int hc    = wgl * HCHUNK;          // first hidden unit owned
  const int tid   = threadIdx.x;
  const int wv    = tid >> 5;
  const int lane  = tid & 31;
  const int li    = lane & 15;
  const int hiL   = lane >> 4;             // 0 for lanes 0-15, 1 for 16-31

  // ---- stage this WG's weight slice into LDS, pre-swizzled into B-fragment order ----
  // frag f = ntile*32 + kk; lane holds column n = ntile*16 + (lane&15),
  // K = kk*32 + (lane<16 ? 0 : 16) .. +16  (16 contiguous bf16 = 32B per lane)
  for (int f = wv; f < 256; f += 8) {
    int ntile = f >> 5, kk = f & 31;
    int nloc  = ntile * 16 + li;
    int g     = nloc >> 5, hu = nloc & 31;
    const unsigned short* src =
        Wc + ((size_t)layer * NGATE + g * HID + hc + hu) * KDIM + kk * 32 + hiL * 16;
    unsigned short* dst = wsl + (size_t)f * 512 + lane * 16;
    *(v8bf*)dst       = *(const v8bf*)src;
    *(v8bf*)(dst + 8) = *(const v8bf*)(src + 8);
  }
  for (int j = tid; j < BATCH * HCHUNK; j += 256) cbuf[j] = 0.0f;
  __syncthreads();

  unsigned* cnt = sync;
  unsigned* gen = sync + 1;

  // pipelined iterations: layer0 handles t = s (s < SEQ); layer1 handles t = s-1 (s >= 1)
  for (int s = 0; s <= SEQ; ++s) {
    const bool active = (layer == 0) ? (s < SEQ) : (s >= 1);
    const int rpar = s & 1, wpar = (s + 1) & 1;

    if (active) {
      const unsigned short* in =
          (layer == 0 ? inb0 : inb1) + (size_t)rpar * BATCH * KDIM;
      const int koffA = hiL ? 16 : 0;
      v8f acc0 = {}, acc1 = {};
      const unsigned short* wp = wsl + (size_t)(wv * 32) * 512;   // wave's 32 frags
      const unsigned short* r0 = in + (size_t)li * KDIM + koffA;         // batch rows 0..15
      const unsigned short* r1 = in + (size_t)(16 + li) * KDIM + koffA;  // batch rows 16..31
#pragma unroll 4
      for (int kk = 0; kk < 32; ++kk) {
        v16bf bfrag = *(const v16bf*)(wp + kk * 512 + lane * 16);
        int kb = kk * 32;
        v16bf a0 = cat16(*(const v8bf*)(r0 + kb), *(const v8bf*)(r0 + kb + 8));
        v16bf a1 = cat16(*(const v8bf*)(r1 + kb), *(const v8bf*)(r1 + kb + 8));
        acc0 = __builtin_amdgcn_wmma_f32_16x16x32_bf16(false, a0, false, bfrag,
                                                       (short)0, acc0, false, false);
        acc1 = __builtin_amdgcn_wmma_f32_16x16x32_bf16(false, a1, false, bfrag,
                                                       (short)0, acc1, false, false);
      }
      // D layout: lane holds column n = lane&15; rows m = r (lanes 0-15) / r+8 (16-31)
#pragma unroll
      for (int r = 0; r < 8; ++r) {
        int m = (hiL ? r + 8 : r);
        gatesbuf[m * NLOC + wv * 16 + li]        = acc0[r];
        gatesbuf[(16 + m) * NLOC + wv * 16 + li] = acc1[r];
      }
    }
    __syncthreads();

    if (active) {
      // elementwise LSTM cell for owned hidden units
      const float* bs = bias + layer * NGATE;
      unsigned short* hb0 = inb0 + (size_t)wpar * BATCH * KDIM;
      unsigned short* hb1 = inb1 + (size_t)wpar * BATCH * KDIM;
      for (int j = tid; j < BATCH * HCHUNK; j += 256) {
        int b = j >> 5, hu = j & 31;
        const float* gr = gatesbuf + b * NLOC;
        float gi = gr[hu]       + bs[0 * HID + hc + hu];
        float gf = gr[32 + hu]  + bs[1 * HID + hc + hu];
        float gg = gr[64 + hu]  + bs[2 * HID + hc + hu];
        float go = gr[96 + hu]  + bs[3 * HID + hc + hu];
        float c  = sigm(gf) * cbuf[j] + sigm(gi) * tanh_fast(gg);
        cbuf[j]  = c;
        float h  = sigm(go) * tanh_fast(c);
        unsigned short hb = f2bf(h);
        int col = hc + hu;
        if (layer == 0) {
          hb0[(size_t)b * KDIM + EMB + col] = hb;  // own recurrence
          hb1[(size_t)b * KDIM + col]       = hb;  // feed layer 1
        } else {
          hb1[(size_t)b * KDIM + EMB + col] = hb;  // own recurrence
          if (s == SEQ) h1f[(size_t)b * HID + col] = h;  // final top-layer h
        }
      }
    }

    // layer-0 WGs gather/convert next timestep's embedding rows (2 batch rows each)
    if (layer == 0 && (s + 1) < SEQ) {
      unsigned short* xdst = inb0 + (size_t)wpar * BATCH * KDIM;
      int b0  = wgl * 2;
      int tk0 = tokens[(size_t)b0 * SEQ + s + 1];
      int tk1 = tokens[(size_t)(b0 + 1) * SEQ + s + 1];
      for (int j = tid; j < 2 * EMB; j += 256) {
        int bb = j >> 9, k = j & 511;
        int tok = bb ? tk1 : tk0;
        xdst[(size_t)(b0 + bb) * KDIM + k] = f2bf(emb[(size_t)tok * EMB + k]);
      }
    }

    grid_sync(cnt, gen);
  }

  // final FC + sigmoid: [32,512] @ [512,2] -> 64 outputs, one WG
  if (wg == NWG - 1 && tid < BATCH * OUTN) {
    int b = tid >> 1, o = tid & 1;
    float acc = fc_b[o];
    const float* hrow = h1f + (size_t)b * HID;
    const float* wrow = fc_w + (size_t)o * HID;
    for (int k = 0; k < HID; ++k) acc += hrow[k] * wrow[k];
    out[b * OUTN + o] = sigm(acc);
  }
}

}  // namespace lstmk

extern "C" void kernel_launch(void* const* d_in, const int* in_sizes, int n_in,
                              void* d_out, int out_size, void* d_ws, size_t ws_size,
                              hipStream_t stream) {
  using namespace lstmk;
  const int*   tokens = (const int*)d_in[0];
  const float* emb    = (const float*)d_in[1];
  const float* w_ih   = (const float*)d_in[2];
  const float* w_hh   = (const float*)d_in[3];
  const float* b_ih   = (const float*)d_in[4];
  const float* b_hh   = (const float*)d_in[5];
  const float* fc_w   = (const float*)d_in[6];
  const float* fc_b   = (const float*)d_in[7];
  float* out = (float*)d_out;

  char* ws = (char*)d_ws;
  unsigned short* Wc   = (unsigned short*)(ws + OFF_WC);
  float*          bias = (float*)(ws + OFF_BIAS);
  unsigned short* inb0 = (unsigned short*)(ws + OFF_IN0);
  unsigned short* inb1 = (unsigned short*)(ws + OFF_IN1);
  float*          h1f  = (float*)(ws + OFF_H1F);
  unsigned*       sync = (unsigned*)(ws + OFF_SYNC);

  prep_weights<<<(2 * NGATE * KDIM + 255) / 256, 256, 0, stream>>>(
      w_ih, w_hh, b_ih, b_hh, Wc, bias);
  prep_state<<<(2 * BATCH * KDIM + 255) / 256, 256, 0, stream>>>(
      tokens, emb, inb0, inb1, sync);
  lstm_persist<<<NWG, 256, SMEM_BYTES, stream>>>(
      tokens, emb, Wc, bias, inb0, inb1, h1f, sync, fc_w, fc_b, out);
}
